// ITDA_64862596104656
// MI455X (gfx1250) — matrix operand order repaired and checked by reading, not compiled, not run on hardware
//
#include <hip/hip_runtime.h>

typedef __attribute__((ext_vector_type(16))) __bf16 v16bf;
typedef __attribute__((ext_vector_type(8)))  float  v8f;

#define B_ROWS 1024
#define DDIM   1024
#define FDIM   32768
#define KSEL   32
#define NCHUNK 32
#define WAVES  8
#define TILES_PER_WAVE (FDIM / 16 / NCHUNK / WAVES)  // 8
#define NT     4                                     // N-tiles processed together per wave
#define SLOTS  (NCHUNK * WAVES)                      // 256
#define SELSTRIDE 34
#define EPSV 1e-3f

struct frag { uint4 q[2]; };  // 32 bytes = one v16bf operand

static __device__ __forceinline__ unsigned short f32_to_bf16(float f) {
  union { float f; unsigned int u; } v; v.f = f;
  unsigned int r = v.u + 0x7FFFu + ((v.u >> 16) & 1u);  // round-to-nearest-even
  return (unsigned short)(r >> 16);
}

// ---------------- one-time conversion of xs to bf16 ----------------
__global__ void k_bf16_convert(const float* __restrict__ src,
                               unsigned short* __restrict__ dst, int n) {
  int i = blockIdx.x * blockDim.x + threadIdx.x;
  int stride = gridDim.x * blockDim.x;
  for (; i < n; i += stride) dst[i] = f32_to_bf16(src[i]);
}

__global__ void k_init(int* __restrict__ nsel) {
  int i = blockIdx.x * blockDim.x + threadIdx.x;
  if (i < B_ROWS) nsel[i] = 0;
}

// ---------------- residual = x - sum_j w_j * xs[idx_j]  (sparse) ----------------
__global__ __launch_bounds__(256) void k_residual(
    const float* __restrict__ x, const float* __restrict__ xs,
    const int* __restrict__ sel_i, const float* __restrict__ sel_w,
    const int* __restrict__ nsel,
    float* __restrict__ resid, unsigned short* __restrict__ resid_bf) {
  int b = blockIdx.x;
  __shared__ int   si[SELSTRIDE];
  __shared__ float sw[SELSTRIDE];
  int m = nsel[b];
  if ((int)threadIdx.x < m) {
    si[threadIdx.x] = sel_i[b * SELSTRIDE + threadIdx.x];
    sw[threadIdx.x] = sel_w[b * SELSTRIDE + threadIdx.x];
  }
  __syncthreads();
  for (int d = threadIdx.x; d < DDIM; d += 256) {
    float r = x[(size_t)b * DDIM + d];
    for (int j = 0; j < m; ++j) r -= sw[j] * xs[(size_t)si[j] * DDIM + d];
    resid[(size_t)b * DDIM + d]    = r;
    resid_bf[(size_t)b * DDIM + d] = f32_to_bf16(r);
  }
}

// ---------------- WMMA bf16 GEMM (residual @ xs^T) fused with row-argmax ----------------
// grid = (B/16, NCHUNK), block = 256 (8 waves). Each wave owns 8 N-tiles of 16 atoms,
// processed in groups of NT=4 with shared A-fragment and double-buffered K pipeline.
__global__ __launch_bounds__(256) void k_scores(
    const unsigned short* __restrict__ xs_bf,
    const unsigned short* __restrict__ resid_bf,
    float* __restrict__ pval, int* __restrict__ pidx) {
  __shared__ unsigned short ldsA[16 * DDIM];  // 32 KB: 16-row residual band, bf16
  int tid = threadIdx.x;
  int rt  = blockIdx.x;

  // Stage the 16x1024 bf16 residual tile into LDS (contiguous 32KB block).
  {
    uint4* l4 = (uint4*)ldsA;
    const uint4* g4 = (const uint4*)(resid_bf + (size_t)rt * 16 * DDIM);
    for (int i = tid; i < (16 * DDIM) / 8; i += 256) l4[i] = g4[i];
  }
  __syncthreads();

  int wave = tid >> 5;
  int lane = tid & 31;
  int lg = lane >> 4;   // lane group (0: lanes 0-15, 1: lanes 16-31)
  int ln = lane & 15;

  float bestv[8]; int besti[8];
#pragma unroll
  for (int r = 0; r < 8; ++r) { bestv[r] = -1e30f; besti[r] = 0; }

  // A-fragment base: row = ln; K chunks at lg*8 and lg*8+16 within each 32-K window
  const unsigned short* ap = ldsA + ln * DDIM + lg * 8;

  for (int s = 0; s < TILES_PER_WAVE; s += NT) {
    int tile0 = blockIdx.y * (WAVES * TILES_PER_WAVE) + wave * TILES_PER_WAVE + s;
    int f0 = tile0 * 16;

    const unsigned short* bp[NT];
#pragma unroll
    for (int j = 0; j < NT; ++j)
      bp[j] = xs_bf + (size_t)(f0 + j * 16 + ln) * DDIM + lg * 8;
    if (s + NT < TILES_PER_WAVE)
      __builtin_prefetch(xs_bf + (size_t)(f0 + NT * 16 + ln) * DDIM + lg * 16, 0, 3);

    // Double-buffered fragments: load K-step k+32 before issuing WMMAs for step k.
    frag A[2], Bf[NT][2];
    A[0].q[0] = *(const uint4*)(ap);
    A[0].q[1] = *(const uint4*)(ap + 16);
#pragma unroll
    for (int j = 0; j < NT; ++j) {
      Bf[j][0].q[0] = *(const uint4*)(bp[j]);
      Bf[j][0].q[1] = *(const uint4*)(bp[j] + 16);
    }

    v8f c[NT];
#pragma unroll
    for (int j = 0; j < NT; ++j) c[j] = {};

#pragma unroll 2
    for (int k0 = 0; k0 < DDIM; k0 += 32) {
      int cur = (k0 >> 5) & 1;
      int nxt = cur ^ 1;
      int kn = k0 + 32;
      if (kn < DDIM) {  // prefetch next K-step fragments into the alternate buffer
        A[nxt].q[0] = *(const uint4*)(ap + kn);
        A[nxt].q[1] = *(const uint4*)(ap + kn + 16);
#pragma unroll
        for (int j = 0; j < NT; ++j) {
          Bf[j][nxt].q[0] = *(const uint4*)(bp[j] + kn);
          Bf[j][nxt].q[1] = *(const uint4*)(bp[j] + kn + 16);
        }
      }
      v16bf va = __builtin_bit_cast(v16bf, A[cur]);
#pragma unroll
      for (int j = 0; j < NT; ++j) {
        v16bf vb = __builtin_bit_cast(v16bf, Bf[j][cur]);
        c[j] = __builtin_amdgcn_wmma_f32_16x16x32_bf16(
                   false, va, false, vb, (short)0, c[j], false, false);
      }
    }

#pragma unroll
    for (int j = 0; j < NT; ++j) {
      int col = f0 + j * 16 + ln;
#pragma unroll
      for (int r = 0; r < 8; ++r) {
        float v = c[j][r];
        if (v > bestv[r]) { bestv[r] = v; besti[r] = col; }
      }
    }
  }

  // Reduce (max, idx) across the 16 lanes that hold the 16 columns of each row.
#pragma unroll
  for (int off = 8; off >= 1; off >>= 1) {
#pragma unroll
    for (int r = 0; r < 8; ++r) {
      float ov = __shfl_xor(bestv[r], off, 32);
      int   oi = __shfl_xor(besti[r], off, 32);
      if (ov > bestv[r] || (ov == bestv[r] && oi < besti[r])) { bestv[r] = ov; besti[r] = oi; }
    }
  }
  if (ln == 0) {
    int slot = blockIdx.y * WAVES + wave;
#pragma unroll
    for (int r = 0; r < 8; ++r) {
      int row = rt * 16 + r + 8 * lg;  // C layout: VGPR r -> M=r (lanes 0-15), M=r+8 (16-31)
      pval[(size_t)row * SLOTS + slot] = bestv[r];
      pidx[(size_t)row * SLOTS + slot] = besti[r];
    }
  }
}

// ---------------- reduce 256 per-row partials to the argmax index ----------------
__global__ __launch_bounds__(256) void k_argmax(
    const float* __restrict__ pval, const int* __restrict__ pidx, int* __restrict__ amax) {
  int b = blockIdx.x, t = threadIdx.x;
  __shared__ float sv[256]; __shared__ int si[256];
  sv[t] = pval[(size_t)b * SLOTS + t];
  si[t] = pidx[(size_t)b * SLOTS + t];
  __syncthreads();
  for (int off = 128; off >= 1; off >>= 1) {
    if (t < off) {
      float ov = sv[t + off]; int oi = si[t + off];
      if (ov > sv[t] || (ov == sv[t] && oi < si[t])) { sv[t] = ov; si[t] = oi; }
    }
    __syncthreads();
  }
  if (t == 0) amax[b] = si[0];
}

// ---------------- per-row pursuit update: grads on selected set, step size, relu ----------------
__global__ __launch_bounds__(256) void k_update(
    const float* __restrict__ xs, const float* __restrict__ resid,
    const int* __restrict__ amax,
    int* __restrict__ sel_i, float* __restrict__ sel_w, int* __restrict__ nsel) {
  int b = blockIdx.x, t = threadIdx.x;
  __shared__ int   si[SELSTRIDE];
  __shared__ float sw[SELSTRIDE];
  __shared__ float g[SELSTRIDE];
  __shared__ float red[256];
  __shared__ int   sm;
  __shared__ float tot_csq, tot_num;

  if (t == 0) {
    int m = nsel[b]; int na = amax[b]; bool found = false;
    for (int j = 0; j < m; ++j) {
      si[j] = sel_i[b * SELSTRIDE + j];
      sw[j] = sel_w[b * SELSTRIDE + j];
      if (si[j] == na) found = true;
    }
    if (!found) { si[m] = na; sw[m] = 0.f; ++m; }
    sm = m;
  }
  __syncthreads();
  int m = sm;
  const float* rrow = resid + (size_t)b * DDIM;

  // g_j = <residual, xs[sel_j]>  (== inner_products at the selected set)
  for (int j = 0; j < m; ++j) {
    const float* arow = xs + (size_t)si[j] * DDIM;
    float p = 0.f;
    for (int d = t; d < DDIM; d += 256) p += rrow[d] * arow[d];
    red[t] = p; __syncthreads();
    for (int off = 128; off >= 1; off >>= 1) { if (t < off) red[t] += red[t + off]; __syncthreads(); }
    if (t == 0) g[j] = red[0];
    __syncthreads();
  }

  // c = grad @ dict ; csq = <c,c> ; num = <c,residual>
  float csq = 0.f, num = 0.f;
  for (int d = t; d < DDIM; d += 256) {
    float cd = 0.f;
    for (int j = 0; j < m; ++j) cd += g[j] * xs[(size_t)si[j] * DDIM + d];
    csq += cd * cd; num += cd * rrow[d];
  }
  red[t] = csq; __syncthreads();
  for (int off = 128; off >= 1; off >>= 1) { if (t < off) red[t] += red[t + off]; __syncthreads(); }
  if (t == 0) tot_csq = red[0];
  __syncthreads();
  red[t] = num; __syncthreads();
  for (int off = 128; off >= 1; off >>= 1) { if (t < off) red[t] += red[t + off]; __syncthreads(); }
  if (t == 0) tot_num = red[0];
  __syncthreads();

  if (t == 0) {
    float step = tot_num / fmaxf(tot_csq, EPSV);
    int mn = 0;
    for (int j = 0; j < m; ++j) {
      float w = sw[j] + step * g[j];
      if (w > 0.f) { sel_i[b * SELSTRIDE + mn] = si[j]; sel_w[b * SELSTRIDE + mn] = w; ++mn; }
    }
    nsel[b] = mn;
  }
}

// ---------------- top-k (sort sparse list), decode x/y, losses ----------------
__global__ __launch_bounds__(256) void k_finalize(
    const float* __restrict__ xs, const float* __restrict__ ys, const float* __restrict__ y,
    const int* __restrict__ sel_i, const float* __restrict__ sel_w, const int* __restrict__ nsel,
    float* __restrict__ out) {
  int b = blockIdx.x, t = threadIdx.x;
  __shared__ int   oi[KSEL];
  __shared__ float ow[KSEL];
  __shared__ float red[256];

  if (t == 0) {
    int m = nsel[b]; if (m > KSEL) m = KSEL;
    int ti[SELSTRIDE]; float tw[SELSTRIDE];
    for (int j = 0; j < m; ++j) { ti[j] = sel_i[b * SELSTRIDE + j]; tw[j] = sel_w[b * SELSTRIDE + j]; }
    for (int a = 0; a < m; ++a) {  // selection sort: desc value, tie -> smaller index
      int bj = a;
      for (int j = a + 1; j < m; ++j)
        if (tw[j] > tw[bj] || (tw[j] == tw[bj] && ti[j] < ti[bj])) bj = j;
      float wv = tw[bj]; int iv = ti[bj];
      tw[bj] = tw[a]; ti[bj] = ti[a]; tw[a] = wv; ti[a] = iv;
    }
    int cand = 0;  // pad with zero-weight smallest unused indices (top_k tie behavior)
    for (int a = m; a < KSEL; ++a) {
      for (;;) {
        bool used = false;
        for (int j = 0; j < m; ++j) if (ti[j] == cand) { used = true; break; }
        if (!used) break;
        ++cand;
      }
      ti[a] = cand; tw[a] = 0.f; ++cand;
    }
    for (int a = 0; a < KSEL; ++a) { oi[a] = ti[a]; ow[a] = tw[a]; }
  }
  __syncthreads();

  float* outW = out;
  float* outI = out + (size_t)B_ROWS * KSEL;
  float* outX = out + (size_t)2 * B_ROWS * KSEL;
  float* outY = outX + (size_t)B_ROWS * DDIM;
  float* outL = outY + (size_t)B_ROWS * DDIM;

  if (t < KSEL) { outW[b * KSEL + t] = ow[t]; outI[b * KSEL + t] = (float)oi[t]; }

  float lp = 0.f;
  for (int d = t; d < DDIM; d += 256) {
    float xr = 0.f, yr = 0.f;
    for (int j = 0; j < KSEL; ++j) {
      float w = ow[j];
      if (w != 0.f) {
        xr += w * xs[(size_t)oi[j] * DDIM + d];
        yr += w * ys[(size_t)oi[j] * DDIM + d];
      }
    }
    outX[(size_t)b * DDIM + d] = xr;
    outY[(size_t)b * DDIM + d] = yr;
    float df = yr - y[(size_t)b * DDIM + d];
    lp += df * df;
  }
  red[t] = lp; __syncthreads();
  for (int off = 128; off >= 1; off >>= 1) { if (t < off) red[t] += red[t + off]; __syncthreads(); }
  if (t == 0) outL[b] = red[0];
}

// ---------------- host side ----------------
extern "C" void kernel_launch(void* const* d_in, const int* in_sizes, int n_in,
                              void* d_out, int out_size, void* d_ws, size_t ws_size,
                              hipStream_t stream) {
  (void)in_sizes; (void)n_in; (void)out_size; (void)ws_size;
  const float* x  = (const float*)d_in[0];
  const float* y  = (const float*)d_in[1];
  const float* xs = (const float*)d_in[2];
  const float* ys = (const float*)d_in[3];

  char* ws = (char*)d_ws;
  size_t off = 0;
  auto alloc = [&](size_t bytes) -> void* {
    void* p = ws + off;
    off = (off + bytes + 255) & ~(size_t)255;
    return p;
  };
  unsigned short* xs_bf    = (unsigned short*)alloc((size_t)FDIM * DDIM * 2);   // 64 MB (L2-hot)
  float*          resid    = (float*)         alloc((size_t)B_ROWS * DDIM * 4); // 4 MB
  unsigned short* resid_bf = (unsigned short*)alloc((size_t)B_ROWS * DDIM * 2); // 2 MB
  float*          pval     = (float*)         alloc((size_t)B_ROWS * SLOTS * 4);
  int*            pidx     = (int*)           alloc((size_t)B_ROWS * SLOTS * 4);
  int*            sel_i    = (int*)           alloc((size_t)B_ROWS * SELSTRIDE * 4);
  float*          sel_w    = (float*)         alloc((size_t)B_ROWS * SELSTRIDE * 4);
  int*            nsel     = (int*)           alloc((size_t)B_ROWS * 4);
  int*            amax     = (int*)           alloc((size_t)B_ROWS * 4);

  k_bf16_convert<<<2048, 256, 0, stream>>>(xs, xs_bf, FDIM * DDIM);
  k_init<<<(B_ROWS + 255) / 256, 256, 0, stream>>>(nsel);

  for (int t = 0; t < KSEL; ++t) {
    k_residual<<<B_ROWS, 256, 0, stream>>>(x, xs, sel_i, sel_w, nsel, resid, resid_bf);
    k_scores<<<dim3(B_ROWS / 16, NCHUNK), 256, 0, stream>>>(xs_bf, resid_bf, pval, pidx);
    k_argmax<<<B_ROWS, 256, 0, stream>>>(pval, pidx, amax);
    k_update<<<B_ROWS, 256, 0, stream>>>(xs, resid, amax, sel_i, sel_w, nsel);
  }
  k_finalize<<<B_ROWS, 256, 0, stream>>>(xs, ys, y, sel_i, sel_w, nsel, (float*)d_out);
}
